// MeshGraphNet_24507083391127
// MI455X (gfx1250) — matrix operand order repaired and checked by reading, not compiled, not run on hardware
//
#include <hip/hip_runtime.h>

// ---------------------------------------------------------------------------
// MeshGraphNet forward for gfx1250 (MI455X), wave32, WMMA f16 16x16x32.
// N=65536 nodes, E=262144 edges, D=128, P=15 steps.
// 32 rows per block (2 m-tiles) so each B fragment feeds 2 WMMAs.
// ---------------------------------------------------------------------------

#define NN 65536
#define EE 262144
#define DD 128
#define PP 15

typedef __attribute__((ext_vector_type(16))) _Float16 v16h;
typedef __attribute__((ext_vector_type(8)))  _Float16 v8h;
typedef __attribute__((ext_vector_type(8)))  float    v8f;

// ---------------------------------------------------------------------------
// WMMA fragment loaders (layouts per CDNA5 ISA 7.12.2, 16-bit A 16x32 / B 32x16)
// ---------------------------------------------------------------------------
__device__ __forceinline__ v16h frag_a(const _Float16* __restrict__ A, int K0,
                                       int kk, int lane) {
  const int m  = lane & 15;
  const int kb = kk * 32 + ((lane & 16) ? 8 : 0);
  const _Float16* p = A + m * K0 + kb;
  v8h lo = *(const v8h*)(p);
  v8h hi = *(const v8h*)(p + 16);
  v16h a;
#pragma unroll
  for (int i = 0; i < 8; ++i) { a[i] = lo[i]; a[i + 8] = hi[i]; }
  return a;
}

__device__ __forceinline__ v16h frag_b(const _Float16* __restrict__ Wt, int K,
                                       int kk, int n, int lane) {
  const int kb = kk * 32 + ((lane & 16) ? 8 : 0);
  const _Float16* p = Wt + (size_t)n * K + kb;
  v8h lo = *(const v8h*)(p);
  v8h hi = *(const v8h*)(p + 16);
  v16h b;
#pragma unroll
  for (int i = 0; i < 8; ++i) { b[i] = lo[i]; b[i + 8] = hi[i]; }
  return b;
}

// ---------------------------------------------------------------------------
// One dense layer: [MT*16 x K0] (LDS f16) @ Wt[K0 x 128] -> MT*16 x 128.
// 8 waves; wave w owns output columns 16w..16w+15 for all MT m-tiles, so each
// B fragment (weight) load is reused MT times.
// ---------------------------------------------------------------------------
template <int K0, int MT>
__device__ __forceinline__ void wmma_layer(const _Float16* __restrict__ A,
                                           const _Float16* __restrict__ Wt,
                                           const float* __restrict__ bias,
                                           bool relu,
                                           _Float16* __restrict__ H16,
                                           float* __restrict__ H32,
                                           int wave, int lane) {
  const int n = wave * 16 + (lane & 15);
  v8f zero = {};
  v8f acc[MT];
#pragma unroll
  for (int mt = 0; mt < MT; ++mt) acc[mt] = zero;
#pragma unroll
  for (int kk = 0; kk < K0 / 32; ++kk) {
    if (kk + 1 < K0 / 32)
      __builtin_prefetch(Wt + (size_t)n * K0 + (kk + 1) * 32, 0, 1);
    v16h b = frag_b(Wt, K0, kk, n, lane);
#pragma unroll
    for (int mt = 0; mt < MT; ++mt) {
      v16h a = frag_a(A + (size_t)mt * 16 * K0, K0, kk, lane);
      acc[mt] = __builtin_amdgcn_wmma_f32_16x16x32_f16(
          false, a, false, b, (short)0, acc[mt], false, false);
    }
  }
  const float bv  = bias[n];
  const int   mhi = (lane & 16) ? 8 : 0;
#pragma unroll
  for (int mt = 0; mt < MT; ++mt) {
#pragma unroll
    for (int r = 0; r < 8; ++r) {
      float v = acc[mt][r] + bv;
      if (relu) v = fmaxf(v, 0.f);
      const int m = mt * 16 + r + mhi;
      if (H16) H16[m * DD + n] = (_Float16)v;
      if (H32) H32[m * DD + n] = v;
    }
  }
}

// ---------------------------------------------------------------------------
// LayerNorm over 128 features per row (+optional residual), store f32 rows.
// ROWS = 16 or 32; 256 threads. resid/out pre-offset to this tile's rows.
// ---------------------------------------------------------------------------
template <int ROWS>
__device__ __forceinline__ void ln_store(const float* __restrict__ sH3,
                                         float* __restrict__ sPS,
                                         float* __restrict__ sPS2,
                                         float* __restrict__ sMu,
                                         float* __restrict__ sRs,
                                         const float* __restrict__ gam,
                                         const float* __restrict__ bet,
                                         const float* __restrict__ resid,
                                         float* __restrict__ out, int t) {
  constexpr int TPR = 256 / ROWS;   // threads per row
  constexpr int CPT = DD / TPR;     // cols per thread
  const int r = t % ROWS, j = t / ROWS;
  float s = 0.f, s2 = 0.f;
#pragma unroll
  for (int i = 0; i < CPT; ++i) {
    float v = sH3[r * DD + j * CPT + i];
    s += v; s2 += v * v;
  }
  sPS[r * (TPR + 1) + j]  = s;
  sPS2[r * (TPR + 1) + j] = s2;
  __syncthreads();
  if (t < ROWS) {
    float a = 0.f, b = 0.f;
    for (int jj = 0; jj < TPR; ++jj) {
      a += sPS[t * (TPR + 1) + jj];
      b += sPS2[t * (TPR + 1) + jj];
    }
    const float mu  = a * (1.f / 128.f);
    const float var = b * (1.f / 128.f) - mu * mu;
    sMu[t] = mu;
    sRs[t] = rsqrtf(var + 1e-5f);
  }
  __syncthreads();
  const float mu = sMu[r], rs = sRs[r];
#pragma unroll
  for (int i = 0; i < CPT; ++i) {
    const int c = j * CPT + i;
    float v = (sH3[r * DD + c] - mu) * rs * gam[c] + bet[c];
    if (resid) v += resid[(size_t)r * DD + c];
    out[(size_t)r * DD + c] = v;
  }
}

// ---------------------------------------------------------------------------
// Weight convert+transpose: W f32 [nmats x K x Nout] -> Wt f16 [nmats x Nout x Kpad]
// ---------------------------------------------------------------------------
__global__ void wcvt_kernel(const float* __restrict__ W, _Float16* __restrict__ Wt,
                            int K, int Kpad, int Nout, int nmats) {
  long i = (long)blockIdx.x * 256 + threadIdx.x;
  const long total = (long)nmats * Nout * Kpad;
  if (i >= total) return;
  const int  s   = (int)(i / ((long)Nout * Kpad));
  const long rem = i - (long)s * Nout * Kpad;
  const int  n = (int)(rem / Kpad), k = (int)(rem % Kpad);
  Wt[(size_t)s * Nout * Kpad + (size_t)n * Kpad + k] =
      (k < K) ? (_Float16)W[(size_t)s * K * Nout + (size_t)k * Nout + n]
              : (_Float16)0.f;
}

// ---------------------------------------------------------------------------
// Encoder: X [M x WIN] f32 -> MLP(WIN->128->128->128)+LN -> out [M x 128] f32
// 32 rows/block.
// ---------------------------------------------------------------------------
template <int WIN>
__global__ __launch_bounds__(256) void encode_kernel(
    const float* __restrict__ X,
    const _Float16* __restrict__ W1t, const float* __restrict__ b1,
    const _Float16* __restrict__ W2t, const float* __restrict__ b2,
    const _Float16* __restrict__ W3t, const float* __restrict__ b3,
    const float* __restrict__ gam, const float* __restrict__ bet,
    float* __restrict__ out) {
  __shared__ alignas(16) _Float16 sA[32 * 32];
  __shared__ alignas(16) _Float16 sH1[32 * DD];
  __shared__ alignas(16) _Float16 sH2[32 * DD];
  __shared__ alignas(16) float    sH3[32 * DD];
  __shared__ float sPS[32 * 9], sPS2[32 * 9], sMu[32], sRs[32];
  const int t = threadIdx.x, lane = t & 31, wave = t >> 5;
  const size_t row0 = (size_t)blockIdx.x * 32;
  for (int idx = t; idx < 32 * 32; idx += 256) {
    const int rr = idx >> 5, c = idx & 31;
    sA[idx] = (c < WIN) ? (_Float16)X[(row0 + rr) * WIN + c] : (_Float16)0.f;
  }
  __syncthreads();
  wmma_layer<32, 2>(sA, W1t, b1, true, sH1, nullptr, wave, lane);
  __syncthreads();
  wmma_layer<DD, 2>(sH1, W2t, b2, true, sH2, nullptr, wave, lane);
  __syncthreads();
  wmma_layer<DD, 2>(sH2, W3t, b3, false, nullptr, sH3, wave, lane);
  __syncthreads();
  ln_store<32>(sH3, sPS, sPS2, sMu, sRs, gam, bet, nullptr, out + row0 * DD, t);
}

// ---------------------------------------------------------------------------
// Edge block: cat(ef, nf[src], nf[dst]) -> MLP(384->128->128->128)+LN + ef
// 32 edges/block; in-place update of ef is safe (rows owned exclusively).
// sH3 (f32, 16KB) aliases sA (24KB), which is dead after layer 1.
// ---------------------------------------------------------------------------
__global__ __launch_bounds__(256) void edge_mlp_kernel(
    float* __restrict__ ef, const float* __restrict__ nf,
    const int* __restrict__ src, const int* __restrict__ dst,
    const _Float16* __restrict__ W1t, const float* __restrict__ b1,
    const _Float16* __restrict__ W2t, const float* __restrict__ b2,
    const _Float16* __restrict__ W3t, const float* __restrict__ b3,
    const float* __restrict__ gam, const float* __restrict__ bet) {
  __shared__ alignas(16) _Float16 sA[32 * 384];   // 24KB; reused as sH3 after L1
  __shared__ alignas(16) _Float16 sH1[32 * DD];
  __shared__ alignas(16) _Float16 sH2[32 * DD];
  __shared__ float sPS[32 * 9], sPS2[32 * 9], sMu[32], sRs[32];
  float* sH3 = (float*)sA;
  const int t = threadIdx.x, lane = t & 31, wave = t >> 5;
  const int r = t & 31;
  const size_t e0 = (size_t)blockIdx.x * 32;
  const size_t e  = e0 + r;
  const int si = src[e], di = dst[e];
  // Stage concat tile [32 x 384] as f16 (nf gathers hit L2: nf is 32MB << 192MB L2)
  for (int cc = (t >> 5); cc < 48; cc += 8) {
    const int c0 = cc * 8;
    const float* sp;
    if (c0 < 128)      sp = ef + e * DD + c0;
    else if (c0 < 256) sp = nf + (size_t)si * DD + (c0 - 128);
    else               sp = nf + (size_t)di * DD + (c0 - 256);
    _Float16* dp = sA + r * 384 + c0;
#pragma unroll
    for (int i = 0; i < 8; ++i) dp[i] = (_Float16)sp[i];
  }
  __syncthreads();
  wmma_layer<384, 2>(sA, W1t, b1, true, sH1, nullptr, wave, lane);
  __syncthreads();
  wmma_layer<DD, 2>(sH1, W2t, b2, true, sH2, nullptr, wave, lane);
  __syncthreads();
  wmma_layer<DD, 2>(sH2, W3t, b3, false, nullptr, sH3, wave, lane);
  __syncthreads();
  ln_store<32>(sH3, sPS, sPS2, sMu, sRs, gam, bet, ef + e0 * DD, ef + e0 * DD, t);
}

// ---------------------------------------------------------------------------
// Node block: cat(nf, agg) -> MLP(256->128->128->128)+LN + nf   (in place)
// 32 nodes/block. sH3 aliases sA (16KB each).
// ---------------------------------------------------------------------------
__global__ __launch_bounds__(256) void node_mlp_kernel(
    float* __restrict__ nf, const float* __restrict__ agg,
    const _Float16* __restrict__ W1t, const float* __restrict__ b1,
    const _Float16* __restrict__ W2t, const float* __restrict__ b2,
    const _Float16* __restrict__ W3t, const float* __restrict__ b3,
    const float* __restrict__ gam, const float* __restrict__ bet) {
  __shared__ alignas(16) _Float16 sA[32 * 256];   // 16KB; reused as sH3 after L1
  __shared__ alignas(16) _Float16 sH1[32 * DD];
  __shared__ alignas(16) _Float16 sH2[32 * DD];
  __shared__ float sPS[32 * 9], sPS2[32 * 9], sMu[32], sRs[32];
  float* sH3 = (float*)sA;
  const int t = threadIdx.x, lane = t & 31, wave = t >> 5;
  const int r = t & 31;
  const size_t n0 = (size_t)blockIdx.x * 32;
  const size_t nd = n0 + r;
  for (int cc = (t >> 5); cc < 32; cc += 8) {
    const int c0 = cc * 8;
    const float* sp = (c0 < 128) ? (nf + nd * DD + c0)
                                 : (agg + nd * DD + (c0 - 128));
    _Float16* dp = sA + r * 256 + c0;
#pragma unroll
    for (int i = 0; i < 8; ++i) dp[i] = (_Float16)sp[i];
  }
  __syncthreads();
  wmma_layer<256, 2>(sA, W1t, b1, true, sH1, nullptr, wave, lane);
  __syncthreads();
  wmma_layer<DD, 2>(sH1, W2t, b2, true, sH2, nullptr, wave, lane);
  __syncthreads();
  wmma_layer<DD, 2>(sH2, W3t, b3, false, nullptr, sH3, wave, lane);
  __syncthreads();
  ln_store<32>(sH3, sPS, sPS2, sMu, sRs, gam, bet, nf + n0 * DD, nf + n0 * DD, t);
}

// ---------------------------------------------------------------------------
// Decoder: nf -> MLP(128->128->128->1), no LN. out [N] f32. 16 rows/block.
// ---------------------------------------------------------------------------
__global__ __launch_bounds__(256) void decode_kernel(
    const float* __restrict__ nf,
    const _Float16* __restrict__ W1t, const float* __restrict__ b1,
    const _Float16* __restrict__ W2t, const float* __restrict__ b2,
    const _Float16* __restrict__ w3, const float* __restrict__ b3,
    float* __restrict__ out) {
  __shared__ alignas(16) _Float16 sA[16 * DD];
  __shared__ alignas(16) _Float16 sH1[16 * DD];
  __shared__ alignas(16) _Float16 sH2[16 * DD];
  __shared__ float sPS[16 * 17];
  const int t = threadIdx.x, lane = t & 31, wave = t >> 5;
  const int r = t & 15, j = t >> 4;
  const size_t n0 = (size_t)blockIdx.x * 16;
  {
    const int c0 = j * 8;
    const float* sp = nf + (n0 + r) * DD + c0;
    _Float16* dp = sA + r * DD + c0;
#pragma unroll
    for (int i = 0; i < 8; ++i) dp[i] = (_Float16)sp[i];
  }
  __syncthreads();
  wmma_layer<DD, 1>(sA, W1t, b1, true, sH1, nullptr, wave, lane);
  __syncthreads();
  wmma_layer<DD, 1>(sH1, W2t, b2, true, sH2, nullptr, wave, lane);
  __syncthreads();
  float s = 0.f;
#pragma unroll
  for (int i = 0; i < 8; ++i) {
    const int c = j * 8 + i;
    s += (float)sH2[r * DD + c] * (float)w3[c];
  }
  sPS[r * 17 + j] = s;
  __syncthreads();
  if (t < 16) {
    float a = 0.f;
    for (int jj = 0; jj < 16; ++jj) a += sPS[t * 17 + jj];
    out[n0 + t] = a + b3[0];
  }
}

// ---------------------------------------------------------------------------
// Scatter-sum helpers
// ---------------------------------------------------------------------------
__global__ void zero_kernel(float* __restrict__ p, int n) {
  const int i = blockIdx.x * 256 + threadIdx.x;
  if (i < n) p[i] = 0.f;
}

__global__ void aggregate_kernel(const float* __restrict__ ef,
                                 const int* __restrict__ dst,
                                 float* __restrict__ agg) {
  const long i = (long)blockIdx.x * 256 + threadIdx.x;  // over E*128
  const int  e = (int)(i >> 7), c = (int)(i & 127);
  atomicAdd(&agg[(size_t)dst[e] * DD + c], ef[i]);
}

// ---------------------------------------------------------------------------
// Host launch. Inputs in setup_inputs() dict (insertion) order; MLP dicts are
// {W1,b1,W2,b2,W3,b3[,g,beta]}.
// ---------------------------------------------------------------------------
extern "C" void kernel_launch(void* const* d_in, const int* in_sizes, int n_in,
                              void* d_out, int out_size, void* d_ws, size_t ws_size,
                              hipStream_t stream) {
  (void)in_sizes; (void)n_in; (void)out_size; (void)ws_size;
  const float* node_features = (const float*)d_in[0];
  const float* edge_features = (const float*)d_in[1];
  const int*   src           = (const int*)d_in[2];
  const int*   dst           = (const int*)d_in[3];
  const float* encn_W1 = (const float*)d_in[4];
  const float* encn_b1 = (const float*)d_in[5];
  const float* encn_W2 = (const float*)d_in[6];
  const float* encn_b2 = (const float*)d_in[7];
  const float* encn_W3 = (const float*)d_in[8];
  const float* encn_b3 = (const float*)d_in[9];
  const float* encn_g  = (const float*)d_in[10];
  const float* encn_be = (const float*)d_in[11];
  const float* ence_W1 = (const float*)d_in[12];
  const float* ence_b1 = (const float*)d_in[13];
  const float* ence_W2 = (const float*)d_in[14];
  const float* ence_b2 = (const float*)d_in[15];
  const float* ence_W3 = (const float*)d_in[16];
  const float* ence_b3 = (const float*)d_in[17];
  const float* ence_g  = (const float*)d_in[18];
  const float* ence_be = (const float*)d_in[19];
  const float* pe_W1 = (const float*)d_in[20];
  const float* pe_b1 = (const float*)d_in[21];
  const float* pe_W2 = (const float*)d_in[22];
  const float* pe_b2 = (const float*)d_in[23];
  const float* pe_W3 = (const float*)d_in[24];
  const float* pe_b3 = (const float*)d_in[25];
  const float* pe_g  = (const float*)d_in[26];
  const float* pe_be = (const float*)d_in[27];
  const float* pn_W1 = (const float*)d_in[28];
  const float* pn_b1 = (const float*)d_in[29];
  const float* pn_W2 = (const float*)d_in[30];
  const float* pn_b2 = (const float*)d_in[31];
  const float* pn_W3 = (const float*)d_in[32];
  const float* pn_b3 = (const float*)d_in[33];
  const float* pn_g  = (const float*)d_in[34];
  const float* pn_be = (const float*)d_in[35];
  const float* dc_W1 = (const float*)d_in[36];
  const float* dc_b1 = (const float*)d_in[37];
  const float* dc_W2 = (const float*)d_in[38];
  const float* dc_b2 = (const float*)d_in[39];
  const float* dc_W3 = (const float*)d_in[40];
  const float* dc_b3 = (const float*)d_in[41];

  char* ws = (char*)d_ws;
  size_t off = 0;
  auto take = [&](size_t bytes) -> char* {
    off = (off + 255) & ~(size_t)255;
    char* p = ws + off;
    off += bytes;
    return p;
  };

  float* nf  = (float*)take((size_t)NN * DD * 4);
  float* ef  = (float*)take((size_t)EE * DD * 4);
  float* agg = (float*)take((size_t)NN * DD * 4);
  _Float16* encn_W1t = (_Float16*)take((size_t)128 * 32 * 2);
  _Float16* encn_W2t = (_Float16*)take((size_t)128 * 128 * 2);
  _Float16* encn_W3t = (_Float16*)take((size_t)128 * 128 * 2);
  _Float16* ence_W1t = (_Float16*)take((size_t)128 * 32 * 2);
  _Float16* ence_W2t = (_Float16*)take((size_t)128 * 128 * 2);
  _Float16* ence_W3t = (_Float16*)take((size_t)128 * 128 * 2);
  _Float16* pe_W1t = (_Float16*)take((size_t)PP * 128 * 384 * 2);
  _Float16* pe_W2t = (_Float16*)take((size_t)PP * 128 * 128 * 2);
  _Float16* pe_W3t = (_Float16*)take((size_t)PP * 128 * 128 * 2);
  _Float16* pn_W1t = (_Float16*)take((size_t)PP * 128 * 256 * 2);
  _Float16* pn_W2t = (_Float16*)take((size_t)PP * 128 * 128 * 2);
  _Float16* pn_W3t = (_Float16*)take((size_t)PP * 128 * 128 * 2);
  _Float16* dc_W1t = (_Float16*)take((size_t)128 * 128 * 2);
  _Float16* dc_W2t = (_Float16*)take((size_t)128 * 128 * 2);
  _Float16* dc_w3t = (_Float16*)take((size_t)128 * 2);

  auto wcvt = [&](const float* W, _Float16* Wt, int K, int Kpad, int Nout, int nm) {
    const long total = (long)nm * Nout * Kpad;
    wcvt_kernel<<<dim3((unsigned)((total + 255) / 256)), dim3(256), 0, stream>>>(
        W, Wt, K, Kpad, Nout, nm);
  };
  wcvt(encn_W1, encn_W1t, 9, 32, 128, 1);
  wcvt(encn_W2, encn_W2t, 128, 128, 128, 1);
  wcvt(encn_W3, encn_W3t, 128, 128, 128, 1);
  wcvt(ence_W1, ence_W1t, 2, 32, 128, 1);
  wcvt(ence_W2, ence_W2t, 128, 128, 128, 1);
  wcvt(ence_W3, ence_W3t, 128, 128, 128, 1);
  wcvt(pe_W1, pe_W1t, 384, 384, 128, PP);
  wcvt(pe_W2, pe_W2t, 128, 128, 128, PP);
  wcvt(pe_W3, pe_W3t, 128, 128, 128, PP);
  wcvt(pn_W1, pn_W1t, 256, 256, 128, PP);
  wcvt(pn_W2, pn_W2t, 128, 128, 128, PP);
  wcvt(pn_W3, pn_W3t, 128, 128, 128, PP);
  wcvt(dc_W1, dc_W1t, 128, 128, 128, 1);
  wcvt(dc_W2, dc_W2t, 128, 128, 128, 1);
  wcvt(dc_W3, dc_w3t, 128, 128, 1, 1);

  // Encoders
  encode_kernel<9><<<dim3(NN / 32), dim3(256), 0, stream>>>(
      node_features, encn_W1t, encn_b1, encn_W2t, encn_b2, encn_W3t, encn_b3,
      encn_g, encn_be, nf);
  encode_kernel<2><<<dim3(EE / 32), dim3(256), 0, stream>>>(
      edge_features, ence_W1t, ence_b1, ence_W2t, ence_b2, ence_W3t, ence_b3,
      ence_g, ence_be, ef);

  // Processor
  for (int s = 0; s < PP; ++s) {
    edge_mlp_kernel<<<dim3(EE / 32), dim3(256), 0, stream>>>(
        ef, nf, src, dst,
        pe_W1t + (size_t)s * 128 * 384, pe_b1 + (size_t)s * 128,
        pe_W2t + (size_t)s * 128 * 128, pe_b2 + (size_t)s * 128,
        pe_W3t + (size_t)s * 128 * 128, pe_b3 + (size_t)s * 128,
        pe_g + (size_t)s * 128, pe_be + (size_t)s * 128);
    zero_kernel<<<dim3((NN * DD) / 256), dim3(256), 0, stream>>>(agg, NN * DD);
    aggregate_kernel<<<dim3((unsigned)(((long)EE * DD) / 256)), dim3(256), 0, stream>>>(
        ef, dst, agg);
    node_mlp_kernel<<<dim3(NN / 32), dim3(256), 0, stream>>>(
        nf, agg,
        pn_W1t + (size_t)s * 128 * 256, pn_b1 + (size_t)s * 128,
        pn_W2t + (size_t)s * 128 * 128, pn_b2 + (size_t)s * 128,
        pn_W3t + (size_t)s * 128 * 128, pn_b3 + (size_t)s * 128,
        pn_g + (size_t)s * 128, pn_be + (size_t)s * 128);
  }

  // Decoder
  decode_kernel<<<dim3(NN / 16), dim3(256), 0, stream>>>(
      nf, dc_W1t, dc_b1, dc_W2t, dc_b2, dc_w3t, dc_b3, (float*)d_out);
}
